// WindowAttentionWav_30356828848465
// MI455X (gfx1250) — matrix operand order repaired and checked by reading, not compiled, not run on hardware
//
#include <hip/hip_runtime.h>

// ---------------------------------------------------------------------------
// CDNA5 (gfx1250) fused window-attention via v_wmma_f32_16x16x32_bf16
// ---------------------------------------------------------------------------

typedef __attribute__((ext_vector_type(16))) __bf16 v16bf;
typedef __attribute__((ext_vector_type(8)))  float  v8f;

#define DIMC   1024
#define NTOK   64
#define NHEAD  8
#define HD     32
#define VD     128
#define SCALE_F 0.17677669529663687f   // 32^-0.5

__device__ __forceinline__ unsigned short f2bf(float f) {
  unsigned u = __builtin_bit_cast(unsigned, f);
  u += 0x7FFFu + ((u >> 16) & 1u);           // round-to-nearest-even
  return (unsigned short)(u >> 16);
}

union FragB16 { v16bf v; uint4 u4[2]; unsigned short s[16]; };

// A fragment: 16x32 bf16, interleaved K (ISA 7.12.2): lane half h ->
//   elems 0..7 = K kb..kb+7, elems 8..15 = K kb+16..kb+23, kb = k0 + 8*h
__device__ __forceinline__ v16bf load_a_bf16(const unsigned short* rowptr, int k0, int half) {
  FragB16 f;
  const int kb = k0 + half * 8;
  f.u4[0] = *(const uint4*)(rowptr + kb);
  f.u4[1] = *(const uint4*)(rowptr + kb + 16);
  return f.v;
}

// B fragment: 32x16 bf16, per-lane column N = lane%16, interleaved K:
//   elems 0..7 = K kb..kb+7, elems 8..15 = K kb+16..kb+23, kb = k0 + 8*half
__device__ __forceinline__ v16bf load_b_bf16(const unsigned short* colptr, int k0, int half) {
  FragB16 f;
  const int kb = k0 + half * 8;
  f.u4[0] = *(const uint4*)(colptr + kb);
  f.u4[1] = *(const uint4*)(colptr + kb + 16);
  return f.v;
}

// ---------------------------------------------------------------------------
// fp32 -> bf16 conversion, 8 elements per thread (vectorized)
// ---------------------------------------------------------------------------
__global__ __launch_bounds__(256)
void f32_to_bf16_v8(const float* __restrict__ src,
                    unsigned short* __restrict__ dst, long n8) {
  const long i = (long)blockIdx.x * blockDim.x + threadIdx.x;
  if (i >= n8) return;
  const float4* s = (const float4*)(src + i * 8);
  const float4 a = s[0], b = s[1];
  unsigned short o[8];
  o[0] = f2bf(a.x); o[1] = f2bf(a.y); o[2] = f2bf(a.z); o[3] = f2bf(a.w);
  o[4] = f2bf(b.x); o[5] = f2bf(b.y); o[6] = f2bf(b.z); o[7] = f2bf(b.w);
  *(uint4*)(dst + i * 8) = *(const uint4*)o;
}

// ---------------------------------------------------------------------------
// fused bias+mask table: bm[w][h][i][j] = rpb[rpi[i][j]][h] + mask[w][i][j]
// total 64*8*64*64 = 2M floats (8 MB, L2-resident, shared by all windows)
// ---------------------------------------------------------------------------
__global__ __launch_bounds__(256)
void build_biasmask(const float* __restrict__ rpb, const int* __restrict__ rpi,
                    const float* __restrict__ mask, float* __restrict__ bm) {
  const int idx = blockIdx.x * 256 + threadIdx.x;      // 0 .. 2M-1
  const int ij = idx & 4095;
  const int h  = (idx >> 12) & 7;
  const int w  = idx >> 15;
  bm[idx] = rpb[rpi[ij] * NHEAD + h] + mask[w * 4096 + ij];
}

// ---------------------------------------------------------------------------
// Y[M][N] = A[M][K] @ W[N][K]^T + bias[N]    (A, W bf16; Y bf16 or f32)
// Block = 256 threads = 8 waves (4Mx2N); block tile 128x128; wave tile 32x64.
// ---------------------------------------------------------------------------
template <bool OUT_BF16>
__global__ __launch_bounds__(256)
void gemm_wmma(const unsigned short* __restrict__ A,
               const unsigned short* __restrict__ W,
               const float* __restrict__ bias, void* __restrict__ Yraw,
               int M, int N, int K) {
  const int lane = threadIdx.x & 31;
  const int wave = threadIdx.x >> 5;
  const int half = lane >> 4;
  const int l16  = lane & 15;
  const int warpM = wave & 3, warpN = wave >> 2;
  const int row0 = blockIdx.x * 128 + warpM * 32;
  const int col0 = blockIdx.y * 128 + warpN * 64;

  const unsigned short* arow[2];
  const unsigned short* wrow[4];
#pragma unroll
  for (int i = 0; i < 2; ++i) arow[i] = A + (long)(row0 + i * 16 + l16) * K;
#pragma unroll
  for (int j = 0; j < 4; ++j) wrow[j] = W + (long)(col0 + j * 16 + l16) * K;

  v8f acc[2][4] = {};

  for (int k0 = 0; k0 < K; k0 += 32) {
    v16bf a[2], b[4];
#pragma unroll
    for (int i = 0; i < 2; ++i) {
      a[i] = load_a_bf16(arow[i], k0, half);
      if (k0 + 32 < K) __builtin_prefetch(arow[i] + k0 + 32, 0, 1);
    }
#pragma unroll
    for (int j = 0; j < 4; ++j) b[j] = load_b_bf16(wrow[j], k0, half);
#pragma unroll
    for (int i = 0; i < 2; ++i)
#pragma unroll
      for (int j = 0; j < 4; ++j)
        acc[i][j] = __builtin_amdgcn_wmma_f32_16x16x32_bf16(
            false, a[i], false, b[j], (short)0, acc[i][j], false, false);
  }

#pragma unroll
  for (int i = 0; i < 2; ++i) {
#pragma unroll
    for (int j = 0; j < 4; ++j) {
      const int c = col0 + j * 16 + l16;
      const float bv = bias[c];
#pragma unroll
      for (int r = 0; r < 8; ++r) {
        const long row = row0 + i * 16 + r + 8 * half;
        const float y = acc[i][j][r] + bv;
        if (OUT_BF16) ((unsigned short*)Yraw)[row * N + c] = f2bf(y);
        else          ((float*)Yraw)[row * N + c] = y;
      }
    }
  }
}

// ---------------------------------------------------------------------------
// Fused attention per (window b, head h):
//   S = scale*(Q K^T) + bm[b%64][h] ; P = softmax(S) ; O = P V
// Block = 128 threads (4 waves); wave w owns query rows 16w..16w+15.
// qk: bf16 [B*64][512] (cols 0..255 Q heads, 256..511 K heads)
// vb: bf16 [B*64][1024]; ctx out: bf16 [B*64][1024]
// ---------------------------------------------------------------------------
__global__ __launch_bounds__(128)
void attn_wmma(const unsigned short* __restrict__ qk,
               const unsigned short* __restrict__ vb,
               const float* __restrict__ biasmask,
               unsigned short* __restrict__ ctx) {
  __shared__ __align__(16) unsigned short vt[VD * NTOK];        // V^T: [dim][token]
  __shared__ __align__(16) unsigned short pbuf[4 * 16 * NTOK];  // per-wave P 16x64

  const int b = blockIdx.x, h = blockIdx.y;
  const int tid  = threadIdx.x;
  const int lane = tid & 31, wave = tid >> 5;
  const int half = lane >> 4, l16 = lane & 15;

  // ---- stage V transposed into LDS ----
  {
    const int t  = tid & 63;
    const int dh = tid >> 6;                           // 0..1 (dim halves)
    const unsigned short* src =
        vb + ((long)(b * NTOK + t)) * DIMC + h * VD + dh * 64;
#pragma unroll
    for (int g = 0; g < 8; ++g) {
      unsigned short tmp[8];
      *(uint4*)tmp = *(const uint4*)(src + g * 8);
#pragma unroll
      for (int e = 0; e < 8; ++e)
        vt[(dh * 64 + g * 8 + e) * NTOK + t] = tmp[e];
    }
  }
  __syncthreads();

  // ---- phase 1: S = Q K^T (4 WMMAs, K=32 = full head dim) ----
  v8f S[4] = {};
  {
    const unsigned short* qrow =
        qk + ((long)(b * NTOK + wave * 16 + l16)) * 512 + h * HD;
    const v16bf a = load_a_bf16(qrow, 0, half);
#pragma unroll
    for (int t = 0; t < 4; ++t) {
      const unsigned short* krow =
          qk + ((long)(b * NTOK + t * 16 + l16)) * 512 + 256 + h * HD;
      const v16bf bf = load_b_bf16(krow, 0, half);
      S[t] = __builtin_amdgcn_wmma_f32_16x16x32_bf16(
          false, a, false, bf, (short)0, S[t], false, false);
    }
  }

  // ---- scale + fused (bias+mask) table (single gather per element) ----
  const float* bm = biasmask + ((long)(b & 63) * NHEAD + h) * (NTOK * NTOK);
#pragma unroll
  for (int t = 0; t < 4; ++t) {
    const int j = t * 16 + l16;
#pragma unroll
    for (int r = 0; r < 8; ++r) {
      const int i = wave * 16 + r + 8 * half;
      S[t][r] = S[t][r] * SCALE_F + bm[i * NTOK + j];
    }
  }

  // ---- row softmax (rows live across one 16-lane half) + P -> LDS ----
  unsigned short* prow = pbuf + wave * (16 * NTOK);
#pragma unroll
  for (int r = 0; r < 8; ++r) {
    float mx = S[0][r];
#pragma unroll
    for (int t = 1; t < 4; ++t) mx = fmaxf(mx, S[t][r]);
#pragma unroll
    for (int off = 8; off >= 1; off >>= 1)
      mx = fmaxf(mx, __shfl_xor(mx, off, 32));
    float e[4], sm = 0.f;
#pragma unroll
    for (int t = 0; t < 4; ++t) { e[t] = __expf(S[t][r] - mx); sm += e[t]; }
#pragma unroll
    for (int off = 8; off >= 1; off >>= 1)
      sm += __shfl_xor(sm, off, 32);
    const float inv = 1.0f / sm;
    const int m = r + 8 * half;
#pragma unroll
    for (int t = 0; t < 4; ++t)
      prow[m * NTOK + t * 16 + l16] = f2bf(e[t] * inv);
  }

  // ---- phase 2: O = P V  (2 K-steps x 8 N-tiles = 16 WMMAs) ----
  v8f O[8] = {};
#pragma unroll
  for (int ks = 0; ks < 2; ++ks) {
    const v16bf a = load_a_bf16(prow + l16 * NTOK, ks * 32, half);
#pragma unroll
    for (int nt = 0; nt < 8; ++nt) {
      const v16bf bf = load_b_bf16(vt + (nt * 16 + l16) * NTOK, ks * 32, half);
      O[nt] = __builtin_amdgcn_wmma_f32_16x16x32_bf16(
          false, a, false, bf, (short)0, O[nt], false, false);
    }
  }

  // ---- store context (bf16) ----
#pragma unroll
  for (int nt = 0; nt < 8; ++nt) {
    const int col = h * VD + nt * 16 + l16;
#pragma unroll
    for (int r = 0; r < 8; ++r) {
      const int i = wave * 16 + r + 8 * half;
      ctx[((long)(b * NTOK + i)) * DIMC + col] = f2bf(O[nt][r]);
    }
  }
}

// ---------------------------------------------------------------------------
extern "C" void kernel_launch(void* const* d_in, const int* in_sizes, int n_in,
                              void* d_out, int out_size, void* d_ws, size_t ws_size,
                              hipStream_t stream) {
  const float* x      = (const float*)d_in[0];
  const int*   rpi    = (const int*)  d_in[1];
  const float* mask   = (const float*)d_in[2];
  const float* qk_w   = (const float*)d_in[3];
  const float* qk_b   = (const float*)d_in[4];
  const float* v_w    = (const float*)d_in[5];
  const float* v_b    = (const float*)d_in[6];
  const float* proj_w = (const float*)d_in[7];
  const float* proj_b = (const float*)d_in[8];
  const float* rpb    = (const float*)d_in[9];

  const long M = 2048L * 64;   // 131072 token rows

  char* ws = (char*)d_ws;
  unsigned short* x_bf   = (unsigned short*)ws; ws += M * 1024 * 2;       // 256 MiB
  unsigned short* qkw_bf = (unsigned short*)ws; ws += 512L * 1024 * 2;
  unsigned short* vw_bf  = (unsigned short*)ws; ws += 1024L * 1024 * 2;
  unsigned short* pw_bf  = (unsigned short*)ws; ws += 1024L * 1024 * 2;
  float*          bm     = (float*)ws;          ws += 64L * 8 * 4096 * 4; // 8 MiB
  unsigned short* qk_out = (unsigned short*)ws; ws += M * 512 * 2;
  unsigned short* v_out  = (unsigned short*)ws; ws += M * 1024 * 2;
  unsigned short* ctx    = (unsigned short*)ws; ws += M * 1024 * 2;

  // one-shot conversions (x converted once instead of per-GEMM-per-N-block)
  f32_to_bf16_v8<<<(M * 1024 / 8 + 255) / 256, 256, 0, stream>>>(x, x_bf, M * 1024 / 8);
  f32_to_bf16_v8<<<(512 * 1024 / 8 + 255) / 256, 256, 0, stream>>>(qk_w, qkw_bf, 512L * 1024 / 8);
  f32_to_bf16_v8<<<(1024 * 1024 / 8 + 255) / 256, 256, 0, stream>>>(v_w, vw_bf, 1024L * 1024 / 8);
  f32_to_bf16_v8<<<(1024 * 1024 / 8 + 255) / 256, 256, 0, stream>>>(proj_w, pw_bf, 1024L * 1024 / 8);

  // fused bias+mask table
  build_biasmask<<<(64 * 8 * 4096) / 256, 256, 0, stream>>>(rpb, rpi, mask, bm);

  // QK projection: [M,1024] x [512,1024]^T -> bf16 [M,512]
  gemm_wmma<true><<<dim3(M / 128, 512 / 128), 256, 0, stream>>>(
      x_bf, qkw_bf, qk_b, qk_out, (int)M, 512, 1024);
  // V projection: [M,1024] x [1024,1024]^T -> bf16 [M,1024]
  gemm_wmma<true><<<dim3(M / 128, 1024 / 128), 256, 0, stream>>>(
      x_bf, vw_bf, v_b, v_out, (int)M, 1024, 1024);

  // fused attention per (window, head)
  attn_wmma<<<dim3(2048, 8), 128, 0, stream>>>(qk_out, v_out, bm, ctx);

  // output projection: bf16 [M,1024] x [1024,1024]^T -> f32 d_out
  gemm_wmma<false><<<dim3(M / 128, 1024 / 128), 256, 0, stream>>>(
      ctx, pw_bf, proj_b, d_out, (int)M, 1024, 1024);
}
